// CompositionalCrossModalReasoning_53188874993990
// MI455X (gfx1250) — compile-verified, matrix operand order checked by reading, not run on hardware
//
#include <hip/hip_runtime.h>

// ---------------------------------------------------------------------------
// CompositionalCrossModalReasoning on MI455X (gfx1250, wave32, WMMA bf16)
// GEMMs via v_wmma_f32_16x16x32_bf16 with ASYNC global->LDS DMA staging
// (global_load_async_to_lds_b128 + s_wait_asynccnt) and LDS double buffering.
// ---------------------------------------------------------------------------

typedef __attribute__((ext_vector_type(16))) __bf16 v16bf;
typedef __attribute__((ext_vector_type(8)))  float  v8f;

#define BB 8
#define LL 512
#define DD 1024
#define HH 16
#define DH 64

__device__ __forceinline__ unsigned short f32_to_bf16_bits(float f) {
  unsigned u = __builtin_bit_cast(unsigned, f);
  unsigned r = u + 0x7fffu + ((u >> 16) & 1u);   // round-to-nearest-even
  return (unsigned short)(r >> 16);
}
__device__ __forceinline__ unsigned pack_bf(float lo, float hi) {
  return (unsigned)f32_to_bf16_bits(lo) | ((unsigned)f32_to_bf16_bits(hi) << 16);
}

// Async DMA: copy 16B from global memory straight into LDS (per lane).
// Tracked by ASYNCcnt; consumed after s_wait_asynccnt + barrier.
__device__ __forceinline__ void async_copy16(unsigned lds_off, const void* gptr) {
  asm volatile("global_load_async_to_lds_b128 %0, %1, off"
               :: "v"(lds_off), "v"((unsigned long long)gptr) : "memory");
}
__device__ __forceinline__ void wait_asynccnt0() {
  asm volatile("s_wait_asynccnt 0x0" ::: "memory");
}

// One 16x32 bf16 WMMA operand fragment from an LDS tile [row][k], row stride
// `rstride` ushorts. CDNA5 16-bit A/B layout: lanes 0-15 hold K 0-7 (VGPR0-3)
// and 16-23 (VGPR4-7); lanes 16-31 hold K 8-15 and 24-31; 2 bf16 per VGPR.
__device__ __forceinline__ v16bf frag_from_tile(const unsigned short* rowbase,
                                                int rstride, int lane) {
  const int half = lane >> 4;
  const unsigned short* row = rowbase + (lane & 15) * rstride;
  v16bf f;
#pragma unroll
  for (int vg = 0; vg < 8; ++vg) {
    const int k = ((vg & 4) << 2) | (half << 3) | ((vg & 3) << 1);
    unsigned u;
    __builtin_memcpy(&u, row + k, 4);            // pairs -> ds_load_b128
    f[2 * vg]     = __builtin_bit_cast(__bf16, (unsigned short)(u & 0xffffu));
    f[2 * vg + 1] = __builtin_bit_cast(__bf16, (unsigned short)(u >> 16));
  }
  return f;
}

// ---------------------------------------------------------------------------
// Batched WMMA GEMM:  C[z] = act(alpha * A[z] @ B[z] + bias)
//   AF32: A is f32 (VGPR-staged, converted to bf16), else bf16 (ASYNC DMA).
//   BNM=1: B stored N-major [N,K] (k contiguous)  -> ASYNC DMA
//   BNM=0: B stored K-major [K,N]                 -> VGPR-staged transpose
//   batch z: offset = (z>>4)*s?h + (z&15)*s?l  (b = z>>4, h = z&15)
//   ACT: 0 none, 1 sigmoid, 2 relu. CF/CB: emit f32 / bf16 outputs.
// Block: 128 threads (4 waves). Block tile 64x64, wave tile 32x32:
// 4 x v_wmma_f32_16x16x32_bf16 per K-step of 32. LDS double-buffered.
// ---------------------------------------------------------------------------
template<int AF32, int BNM, int ACT, int BIAS, int HASCF, int HASCB>
__global__ __launch_bounds__(128) void gemm_t(
    const void* __restrict__ Ap, int lda, long long sAh, long long sAl,
    const unsigned short* __restrict__ Bp0, int ldb, long long sBh, long long sBl,
    const float* __restrict__ bias, float alpha,
    float* __restrict__ Cfp, int ldcf, long long sCfh, long long sCfl,
    unsigned short* __restrict__ Cbp, int ldcb, long long sCbh, long long sCbl,
    int K) {
  __shared__ unsigned short As[2][64][40];   // [buf][m][k], stride 40 (16B rows)
  __shared__ unsigned short Bs[2][64][40];   // [buf][n][k]

  constexpr bool ASY = (!AF32) || (BNM == 1);   // any async traffic?

  const int t  = threadIdx.x;
  const int z  = blockIdx.z;
  const long long zh = z >> 4, zl = z & 15;
  const int m0 = blockIdx.y << 6;
  const int n0 = blockIdx.x << 6;

  const float*          Af = (const float*)Ap + (AF32 ? (zh * sAh + zl * sAl) : 0);
  const unsigned short* Ab = (const unsigned short*)Ap + (AF32 ? 0 : (zh * sAh + zl * sAl));
  const unsigned short* Bp = Bp0 + zh * sBh + zl * sBl;

  // tile-staging thread mapping: 16 contiguous elems (32B bf16) per thread
  const int arow = t >> 1, acb = (t & 1) << 4;   // 64 rows x 32 k
  const int bkk  = t & 31, bng = (t >> 5) << 4;  // BNM=0 path: 32 k x 64 n

  float4 fa0, fa1, fa2, fa3;                     // AF32 staging
  uint4  rb0, rb1;                               // BNM=0 staging

  // Issue phase: async DMA straight to LDS, or global loads into regs.
  auto issueA = [&](int k0, int buf) {
    if constexpr (AF32) {
      const float* p = Af + (long long)(m0 + arow) * lda + k0 + acb;
      fa0 = *(const float4*)(p);     fa1 = *(const float4*)(p + 4);
      fa2 = *(const float4*)(p + 8); fa3 = *(const float4*)(p + 12);
      __builtin_prefetch((const void*)(p + 32), 0, 1);   // global_prefetch_b8
    } else {
      const unsigned short* p = Ab + (long long)(m0 + arow) * lda + k0 + acb;
      const unsigned d = (unsigned)(size_t)(&As[buf][arow][acb]);
      async_copy16(d, p);
      async_copy16(d + 16u, p + 8);
    }
  };
  auto issueB = [&](int k0, int buf) {
    if constexpr (BNM) {
      const unsigned short* p = Bp + (long long)(n0 + arow) * ldb + k0 + acb;
      const unsigned d = (unsigned)(size_t)(&Bs[buf][arow][acb]);
      async_copy16(d, p);
      async_copy16(d + 16u, p + 8);
    } else {
      const unsigned short* p = Bp + (long long)(k0 + bkk) * ldb + n0 + bng;
      rb0 = *(const uint4*)(p); rb1 = *(const uint4*)(p + 8);
      __builtin_prefetch((const void*)(p + (long long)32 * ldb), 0, 1);
    }
  };
  // Flush phase: reg-staged operands written to LDS (after current WMMAs).
  auto flushA = [&](int buf) {
    if constexpr (AF32) {
      unsigned short* d = &As[buf][arow][acb];
      uint4 u0 = { pack_bf(fa0.x, fa0.y), pack_bf(fa0.z, fa0.w),
                   pack_bf(fa1.x, fa1.y), pack_bf(fa1.z, fa1.w) };
      uint4 u1 = { pack_bf(fa2.x, fa2.y), pack_bf(fa2.z, fa2.w),
                   pack_bf(fa3.x, fa3.y), pack_bf(fa3.z, fa3.w) };
      *(uint4*)d = u0; *(uint4*)(d + 8) = u1;              // ds_store_b128 x2
    }
  };
  auto flushB = [&](int buf) {
    if constexpr (!BNM) {
      union { uint4 v[2]; unsigned short s[16]; } u;
      u.v[0] = rb0; u.v[1] = rb1;
#pragma unroll
      for (int j = 0; j < 16; ++j) Bs[buf][bng + j][bkk] = u.s[j];  // transpose
    }
  };

  const int w = t >> 5, lane = t & 31;
  const int wm = (w & 1) << 5;     // wave row offset: 0/32
  const int wn = (w >> 1) << 5;    // wave col offset: 0/32

  const v8f vzero = {0.f, 0.f, 0.f, 0.f, 0.f, 0.f, 0.f, 0.f};
  v8f c00 = vzero, c01 = vzero, c10 = vzero, c11 = vzero;

  issueA(0, 0); issueB(0, 0);
  flushA(0);    flushB(0);
  if constexpr (ASY) wait_asynccnt0();
  __syncthreads();

  int cur = 0;
  for (int k0 = 0; k0 < K; k0 += 32) {
    const bool nxt = (k0 + 32) < K;
    if (nxt) { issueA(k0 + 32, cur ^ 1); issueB(k0 + 32, cur ^ 1); }  // DMA in flight
    const v16bf a0 = frag_from_tile(&As[cur][wm][0],      40, lane);
    const v16bf a1 = frag_from_tile(&As[cur][wm + 16][0], 40, lane);
    const v16bf b0 = frag_from_tile(&Bs[cur][wn][0],      40, lane);
    const v16bf b1 = frag_from_tile(&Bs[cur][wn + 16][0], 40, lane);
    c00 = __builtin_amdgcn_wmma_f32_16x16x32_bf16(false, a0, false, b0, (short)0, c00, false, false);
    c01 = __builtin_amdgcn_wmma_f32_16x16x32_bf16(false, a0, false, b1, (short)0, c01, false, false);
    c10 = __builtin_amdgcn_wmma_f32_16x16x32_bf16(false, a1, false, b0, (short)0, c10, false, false);
    c11 = __builtin_amdgcn_wmma_f32_16x16x32_bf16(false, a1, false, b1, (short)0, c11, false, false);
    if (nxt) { flushA(cur ^ 1); flushB(cur ^ 1); }
    if constexpr (ASY) wait_asynccnt0();      // my DMA landed before barrier
    __syncthreads();
    cur ^= 1;
  }

  // C/D layout: VGPR r -> M = r + 8*(lane>=16); N = lane&15
  const int half = lane >> 4, nl = lane & 15;
  const long long cfo = zh * sCfh + zl * sCfl;
  const long long cbo = zh * sCbh + zl * sCbl;
#define EPI(CC, AI, BI)                                                         \
  {                                                                             \
    _Pragma("unroll")                                                           \
    for (int r = 0; r < 8; ++r) {                                               \
      const int m = m0 + wm + (AI << 4) + r + (half << 3);                      \
      const int n = n0 + wn + (BI << 4) + nl;                                   \
      float v = CC[r] * alpha;                                                  \
      if constexpr (BIAS) v += bias[n];                                         \
      if constexpr (ACT == 1) v = 1.0f / (1.0f + __expf(-v));                   \
      if constexpr (ACT == 2) v = fmaxf(v, 0.0f);                               \
      if constexpr (HASCF) Cfp[cfo + (long long)m * ldcf + n] = v;              \
      if constexpr (HASCB) Cbp[cbo + (long long)m * ldcb + n] = f32_to_bf16_bits(v); \
    }                                                                           \
  }
  EPI(c00, 0, 0) EPI(c01, 0, 1) EPI(c10, 1, 0) EPI(c11, 1, 1)
#undef EPI
}

// ---------------------------------------------------------------------------
// Elementwise helpers (all 128-bit vectorized)
// ---------------------------------------------------------------------------
extern "C" __global__ void __launch_bounds__(256)
cvt_bf16(const float* __restrict__ src, unsigned short* __restrict__ dst, int n4) {
  const int i = blockIdx.x * 256 + threadIdx.x;
  if (i >= n4) return;
  const float4 f = ((const float4*)src)[i];
  uint2 o; o.x = pack_bf(f.x, f.y); o.y = pack_bf(f.z, f.w);
  ((uint2*)dst)[i] = o;
}

// src [K,N] f32 -> dst [N,K] bf16 (N-major weights for the GEMM B operand)
extern "C" __global__ void __launch_bounds__(256)
transpose_bf16(const float* __restrict__ src, unsigned short* __restrict__ dst,
               int K, int N) {
  __shared__ unsigned short tile[32][33];
  const int bx = blockIdx.x * 32;              // n base
  const int by = blockIdx.y * 32;              // k base
  const int tx = threadIdx.x & 31, ty = threadIdx.x >> 5;
#pragma unroll
  for (int r = ty; r < 32; r += 8)
    tile[r][tx] = f32_to_bf16_bits(src[(long long)(by + r) * N + bx + tx]);
  __syncthreads();
#pragma unroll
  for (int r = ty; r < 32; r += 8)
    dst[(long long)(bx + r) * K + by + tx] = tile[tx][r];
}

// wave-per-row in-place softmax over 512 cols; row held in registers
extern "C" __global__ void __launch_bounds__(256)
softmax_rows(float* __restrict__ p, int rows) {
  const int wave = threadIdx.x >> 5, lane = threadIdx.x & 31;
  const int row = blockIdx.x * 8 + wave;
  if (row >= rows) return;
  float4* r = (float4*)(p + (long long)row * LL);   // 128 float4
  float4 v[4];
  float mx = -3.4e38f;
#pragma unroll
  for (int i = 0; i < 4; ++i) {
    v[i] = r[lane + i * 32];
    mx = fmaxf(mx, fmaxf(fmaxf(v[i].x, v[i].y), fmaxf(v[i].z, v[i].w)));
  }
#pragma unroll
  for (int o = 16; o > 0; o >>= 1) mx = fmaxf(mx, __shfl_xor(mx, o, 32));
  float s = 0.0f;
#pragma unroll
  for (int i = 0; i < 4; ++i) {
    v[i].x = __expf(v[i].x - mx); v[i].y = __expf(v[i].y - mx);
    v[i].z = __expf(v[i].z - mx); v[i].w = __expf(v[i].w - mx);
    s += v[i].x + v[i].y + v[i].z + v[i].w;
  }
#pragma unroll
  for (int o = 16; o > 0; o >>= 1) s += __shfl_xor(s, o, 32);
  const float inv = 1.0f / s;
#pragma unroll
  for (int i = 0; i < 4; ++i) {
    v[i].x *= inv; v[i].y *= inv; v[i].z *= inv; v[i].w *= inv;
    r[lane + i * 32] = v[i];
  }
}

// avg over heads: attn [B*H,L,L] f32 -> out [B,L,L] f32 ; n4 = B*L*L/4
extern "C" __global__ void __launch_bounds__(256)
avg_heads(const float* __restrict__ attn, float* __restrict__ out, int n4) {
  const int i = blockIdx.x * 256 + threadIdx.x;
  if (i >= n4) return;
  const int b = i >> 16;                       // / (512*512/4)
  const int ij = i & 65535;
  const float4* p = (const float4*)attn + (long long)b * HH * 65536 + ij;
  float4 s = {0.f, 0.f, 0.f, 0.f};
#pragma unroll
  for (int h = 0; h < HH; ++h) {
    const float4 x = p[(long long)h * 65536];
    s.x += x.x; s.y += x.y; s.z += x.z; s.w += x.w;
  }
  const float k = 1.0f / (float)HH;
  s.x *= k; s.y *= k; s.z *= k; s.w *= k;
  ((float4*)out)[i] = s;
}

// concat [text | image | text_cross] -> [rows, 3072] bf16, 8 elems/thread
extern "C" __global__ void __launch_bounds__(256)
pack_concat(const uint4* __restrict__ tx, const uint4* __restrict__ im,
            const uint4* __restrict__ cr, uint4* __restrict__ out, int rows) {
  const int i = blockIdx.x * 256 + threadIdx.x;
  const int total = rows * 384;                // 3072/8 groups per row
  if (i >= total) return;
  const int r = i / 384, c = i - r * 384;
  uint4 v;
  if (c < 128)      v = tx[r * 128 + c];
  else if (c < 256) v = im[r * 128 + (c - 128)];
  else              v = cr[r * 128 + (c - 256)];
  out[i] = v;
}

// ---------------------------------------------------------------------------
// Host-side orchestration. Workspace = 256 MB; the 128 MB f32 attention
// tensor stays L2-resident (192 MB) across scores->softmax->avg->AV.
// ---------------------------------------------------------------------------
#define OFF_XT     0ll
#define OFF_XI     8388608ll
#define OFF_WQ     16777216ll
#define OFF_WK     18874368ll
#define OFF_WV     20971520ll
#define OFF_WO     23068672ll
#define OFF_W1     25165824ll
#define OFF_W2     31457280ll
#define OFF_MUQT   33554432ll
#define OFF_MUKT   41943040ll
#define OFF_VT     50331648ll
#define OFF_MUQI   58720256ll
#define OFF_MUKI   67108864ll
#define OFF_VI     75497472ll
#define OFF_SCORES 83886080ll
#define OFF_CTX    218103808ll
#define OFF_CROSS  226492416ll
#define OFF_CC     234881024ll
#define OFF_HID    260046848ll

extern "C" void kernel_launch(void* const* d_in, const int* in_sizes, int n_in,
                              void* d_out, int out_size, void* d_ws, size_t ws_size,
                              hipStream_t stream) {
  const float* text  = (const float*)d_in[0];
  const float* image = (const float*)d_in[1];
  const float* Wq = (const float*)d_in[2];  const float* bq = (const float*)d_in[3];
  const float* Wk = (const float*)d_in[4];  const float* bk = (const float*)d_in[5];
  const float* Wv = (const float*)d_in[6];  const float* bv = (const float*)d_in[7];
  const float* Wo = (const float*)d_in[8];  const float* bo = (const float*)d_in[9];
  const float* W1 = (const float*)d_in[10]; const float* b1 = (const float*)d_in[11];
  const float* W2 = (const float*)d_in[12]; const float* b2 = (const float*)d_in[13];

  char* ws = (char*)d_ws;
  unsigned short* Xt   = (unsigned short*)(ws + OFF_XT);
  unsigned short* Xi   = (unsigned short*)(ws + OFF_XI);
  unsigned short* WqT  = (unsigned short*)(ws + OFF_WQ);   // [N,K] bf16
  unsigned short* WkT  = (unsigned short*)(ws + OFF_WK);
  unsigned short* WvT  = (unsigned short*)(ws + OFF_WV);
  unsigned short* WoT  = (unsigned short*)(ws + OFF_WO);
  unsigned short* W1T  = (unsigned short*)(ws + OFF_W1);   // [1024,3072]
  unsigned short* W2T  = (unsigned short*)(ws + OFF_W2);
  unsigned short* muQt = (unsigned short*)(ws + OFF_MUQT);
  unsigned short* muKt = (unsigned short*)(ws + OFF_MUKT);
  unsigned short* Vt   = (unsigned short*)(ws + OFF_VT);
  unsigned short* muQi = (unsigned short*)(ws + OFF_MUQI);
  unsigned short* muKi = (unsigned short*)(ws + OFF_MUKI);
  unsigned short* Vi   = (unsigned short*)(ws + OFF_VI);
  float*          sc   = (float*)(ws + OFF_SCORES);
  unsigned short* ctx  = (unsigned short*)(ws + OFF_CTX);
  unsigned short* crs  = (unsigned short*)(ws + OFF_CROSS);
  unsigned short* cc   = (unsigned short*)(ws + OFF_CC);
  unsigned short* hid  = (unsigned short*)(ws + OFF_HID);

  float* out_tc  = (float*)d_out;                 // text_cross   [8,512,1024]
  float* out_ic  = out_tc + 4194304;              // image_cross  [8,512,1024]
  float* out_cp  = out_tc + 8388608;              // compositional[8,512,1024]
  float* out_avg = out_tc + 12582912;             // avg_attn     [8,512,512]

  const int MR = BB * LL;                         // 4096
  const dim3 b256(256), b128(128);

  // ---- 1) features -> bf16; weights -> transposed bf16 ----
  cvt_bf16<<<dim3(MR * DD / 4 / 256), b256, 0, stream>>>(text,  Xt, MR * DD / 4);
  cvt_bf16<<<dim3(MR * DD / 4 / 256), b256, 0, stream>>>(image, Xi, MR * DD / 4);
  transpose_bf16<<<dim3(DD / 32, DD / 32),     b256, 0, stream>>>(Wq, WqT, DD, DD);
  transpose_bf16<<<dim3(DD / 32, DD / 32),     b256, 0, stream>>>(Wk, WkT, DD, DD);
  transpose_bf16<<<dim3(DD / 32, DD / 32),     b256, 0, stream>>>(Wv, WvT, DD, DD);
  transpose_bf16<<<dim3(DD / 32, DD / 32),     b256, 0, stream>>>(Wo, WoT, DD, DD);
  transpose_bf16<<<dim3(DD / 32, 3 * DD / 32), b256, 0, stream>>>(W1, W1T, 3 * DD, DD);
  transpose_bf16<<<dim3(DD / 32, DD / 32),     b256, 0, stream>>>(W2, W2T, DD, DD);

  // ---- 2) six projection GEMMs ----
  const dim3 gproj(DD / 64, MR / 64, 1);
  gemm_t<0,1,1,1,0,1><<<gproj, b128, 0, stream>>>((const void*)Xt, DD, 0ll, 0ll,
      WqT, DD, 0ll, 0ll, bq, 1.0f, nullptr, 0, 0ll, 0ll, muQt, DD, 0ll, 0ll, DD);
  gemm_t<0,1,1,1,0,1><<<gproj, b128, 0, stream>>>((const void*)Xt, DD, 0ll, 0ll,
      WkT, DD, 0ll, 0ll, bk, 1.0f, nullptr, 0, 0ll, 0ll, muKt, DD, 0ll, 0ll, DD);
  gemm_t<0,1,0,1,0,1><<<gproj, b128, 0, stream>>>((const void*)Xt, DD, 0ll, 0ll,
      WvT, DD, 0ll, 0ll, bv, 1.0f, nullptr, 0, 0ll, 0ll, Vt, DD, 0ll, 0ll, DD);
  gemm_t<0,1,1,1,0,1><<<gproj, b128, 0, stream>>>((const void*)Xi, DD, 0ll, 0ll,
      WqT, DD, 0ll, 0ll, bq, 1.0f, nullptr, 0, 0ll, 0ll, muQi, DD, 0ll, 0ll, DD);
  gemm_t<0,1,1,1,0,1><<<gproj, b128, 0, stream>>>((const void*)Xi, DD, 0ll, 0ll,
      WkT, DD, 0ll, 0ll, bk, 1.0f, nullptr, 0, 0ll, 0ll, muKi, DD, 0ll, 0ll, DD);
  gemm_t<0,1,0,1,0,1><<<gproj, b128, 0, stream>>>((const void*)Xi, DD, 0ll, 0ll,
      WvT, DD, 0ll, 0ll, bv, 1.0f, nullptr, 0, 0ll, 0ll, Vi, DD, 0ll, 0ll, DD);

  const long long sBH = (long long)LL * DD;   // (b) stride in mu/V
  const long long sHH = (long long)DH;        // (h) stride
  const long long sSB = 16ll * LL * LL;       // (b) stride in scores
  const long long sSH = (long long)LL * LL;   // (h) stride
  const dim3 gsc(LL / 64, LL / 64, BB * HH);
  const dim3 gav(1, LL / 64, BB * HH);
  const dim3 gout(DD / 64, MR / 64, 1);

  // ================= text direction (Q=text, K/V=image) =================
  gemm_t<0,1,0,0,1,0><<<gsc, b128, 0, stream>>>((const void*)muQt, DD, sBH, sHH,
      muKi, DD, sBH, sHH, nullptr, 1.0f / (float)DH,
      sc, LL, sSB, sSH, nullptr, 0, 0ll, 0ll, DH);
  softmax_rows<<<dim3(BB * HH * LL / 8), b256, 0, stream>>>(sc, BB * HH * LL);
  avg_heads<<<dim3(BB * LL * LL / 4 / 256), b256, 0, stream>>>(sc, out_avg, BB * LL * LL / 4);
  gemm_t<1,0,0,0,0,1><<<gav, b128, 0, stream>>>((const void*)sc, LL, sSB, sSH,
      Vi, DD, sBH, sHH, nullptr, 1.0f,
      nullptr, 0, 0ll, 0ll, ctx, DD, sBH, sHH, LL);
  gemm_t<0,1,0,1,1,1><<<gout, b128, 0, stream>>>((const void*)ctx, DD, 0ll, 0ll,
      WoT, DD, 0ll, 0ll, bo, 1.0f,
      out_tc, DD, 0ll, 0ll, crs, DD, 0ll, 0ll, DD);

  // ================= image direction (Q=image, K/V=text) =================
  gemm_t<0,1,0,0,1,0><<<gsc, b128, 0, stream>>>((const void*)muQi, DD, sBH, sHH,
      muKt, DD, sBH, sHH, nullptr, 1.0f / (float)DH,
      sc, LL, sSB, sSH, nullptr, 0, 0ll, 0ll, DH);
  softmax_rows<<<dim3(BB * HH * LL / 8), b256, 0, stream>>>(sc, BB * HH * LL);
  gemm_t<1,0,0,0,0,1><<<gav, b128, 0, stream>>>((const void*)sc, LL, sSB, sSH,
      Vt, DD, sBH, sHH, nullptr, 1.0f,
      nullptr, 0, 0ll, 0ll, ctx, DD, sBH, sHH, LL);
  gemm_t<0,1,0,1,1,0><<<gout, b128, 0, stream>>>((const void*)ctx, DD, 0ll, 0ll,
      WoT, DD, 0ll, 0ll, bo, 1.0f,
      out_ic, DD, 0ll, 0ll, nullptr, 0, 0ll, 0ll, DD);

  // ================= compositional FFN =================
  pack_concat<<<dim3(MR * 384 / 256), b256, 0, stream>>>(
      (const uint4*)Xt, (const uint4*)Xi, (const uint4*)crs, (uint4*)cc, MR);
  gemm_t<0,1,2,1,0,1><<<gout, b128, 0, stream>>>((const void*)cc, 3 * DD, 0ll, 0ll,
      W1T, 3 * DD, 0ll, 0ll, b1, 1.0f,
      nullptr, 0, 0ll, 0ll, hid, DD, 0ll, 0ll, 3 * DD);
  gemm_t<0,1,0,1,1,0><<<gout, b128, 0, stream>>>((const void*)hid, DD, 0ll, 0ll,
      W2T, DD, 0ll, 0ll, b2, 1.0f,
      out_cp, DD, 0ll, 0ll, nullptr, 0, 0ll, 0ll, DD);

  (void)in_sizes; (void)n_in; (void)out_size; (void)ws_size;
}
